// HASNN_36653250904180
// MI455X (gfx1250) — matrix operand order repaired and checked by reading, not compiled, not run on hardware
//
#include <hip/hip_runtime.h>
#include <math.h>

// ---------------- problem constants ----------------
#define T_STEPS 8
#define N_NODES 100000
#define F_DIM   128
#define B_SEEDS 4096
#define S1_     5
#define S2_     2
#define H1_     512
#define H2_     64
#define O_      32
#define C_      2

#define M1 (B_SEEDS + B_SEEDS * S1_)   // 24576 rows in layer-1 GEMM
#define K1 (2 * F_DIM)                 // 256  (self || nbr)
#define N1 H1_                         // 512
#define M2 B_SEEDS                     // 4096
#define K2 (2 * H1_)                   // 1024 (s0 || agg)
#define N2 H2_                         // 64

#define PITCH1 (K1 + 8)                // LDS row pitch: 256 + 8 elems (16B pad)

typedef __attribute__((ext_vector_type(16))) __bf16 v16bf;
typedef __attribute__((ext_vector_type(8)))  __bf16 v8bf;
typedef __attribute__((ext_vector_type(4)))  __bf16 v4bf;
typedef __attribute__((ext_vector_type(8)))  float  v8f;

#if defined(__HIP_DEVICE_COMPILE__) && __has_builtin(__builtin_amdgcn_tensor_load_to_lds)
#define USE_TDM 1
#else
#define USE_TDM 0
#endif

// ---------------- weight prep: W^T bf16, self||nbr concat along K --------
__global__ __launch_bounds__(256)
void k_prep_w1(const float* __restrict__ Ws1, const float* __restrict__ Wn1,
               __bf16* __restrict__ WT1) {
  int idx = blockIdx.x * 256 + threadIdx.x;          // [c][n][k], k fastest
  if (idx >= C_ * N1 * K1) return;
  int k = idx & (K1 - 1);
  int n = (idx >> 8) & (N1 - 1);
  int c = idx >> 17;
  float v = (k < F_DIM)
      ? Ws1[((size_t)c * F_DIM + k) * H1_ + n]
      : Wn1[((size_t)c * F_DIM + (k - F_DIM)) * H1_ + n];
  WT1[idx] = (__bf16)v;
}

__global__ __launch_bounds__(256)
void k_prep_w2(const float* __restrict__ Ws2, const float* __restrict__ Wn2,
               __bf16* __restrict__ WT2) {
  int idx = blockIdx.x * 256 + threadIdx.x;          // [c][n][k], k fastest
  if (idx >= C_ * N2 * K2) return;
  int k = idx & (K2 - 1);
  int n = (idx >> 10) & (N2 - 1);
  int c = idx >> 16;
  float v = (k < H1_)
      ? Ws2[((size_t)c * H1_ + k) * H2_ + n]
      : Wn2[((size_t)c * H1_ + (k - H1_)) * H2_ + n];
  WT2[idx] = (__bf16)v;
}

// ---------- gather + neighbor mean + pack to bf16: Abuf[M1][K1] ----------
__global__ __launch_bounds__(256)
void k_build_A(const float* __restrict__ xt, const int* __restrict__ nodes,
               const int* __restrict__ hop1, const int* __restrict__ hop2,
               __bf16* __restrict__ Abuf) {
  int wave = threadIdx.x >> 5, lane = threadIdx.x & 31;
  int row = blockIdx.x * 8 + wave;                   // 0..M1-1
  const float4* xv = (const float4*)xt;              // 32 float4 per node row
  float4 self;
  float sx = 0.f, sy = 0.f, sz = 0.f, sw = 0.f, scale;
  if (row < B_SEEDS) {
    self = xv[(size_t)nodes[row] * 32 + lane];
#pragma unroll
    for (int j = 0; j < S1_; ++j) {
      float4 v = xv[(size_t)hop1[row * S1_ + j] * 32 + lane];
      sx += v.x; sy += v.y; sz += v.z; sw += v.w;
    }
    scale = 1.0f / S1_;
  } else {
    int i = row - B_SEEDS;
    self = xv[(size_t)hop1[i] * 32 + lane];
#pragma unroll
    for (int j = 0; j < S2_; ++j) {
      float4 v = xv[(size_t)hop2[i * S2_ + j] * 32 + lane];
      sx += v.x; sy += v.y; sz += v.z; sw += v.w;
    }
    scale = 1.0f / S2_;
  }
  __bf16* orow = Abuf + (size_t)row * K1;
  int col = lane * 4;
  v4bf s; s[0] = (__bf16)self.x; s[1] = (__bf16)self.y;
          s[2] = (__bf16)self.z; s[3] = (__bf16)self.w;
  *(v4bf*)(orow + col) = s;
  v4bf a; a[0] = (__bf16)(sx * scale); a[1] = (__bf16)(sy * scale);
          a[2] = (__bf16)(sz * scale); a[3] = (__bf16)(sw * scale);
  *(v4bf*)(orow + F_DIM + col) = a;
}

// ---------------- TDM: 2D tile (64 rows x 512B) global -> LDS -----------
// D# per CDNA5 ISA ch.8: group0 = {flags, lds_addr, global_addr, type=2},
// group1 = {data_size=2B, pad 16B per 512B row, dims/strides}.
#if USE_TDM
__device__ __forceinline__
void tdm_load_B1(unsigned lds_off, const void* gsrc) {
  typedef __attribute__((ext_vector_type(4))) unsigned int tdm_v4u;
  typedef __attribute__((ext_vector_type(8))) int          tdm_v8i;
  typedef __attribute__((ext_vector_type(4))) int          tdm_v4i;
  unsigned long long ga = (unsigned long long)(size_t)gsrc;
  tdm_v4u g0;
  g0[0] = 1u;                                        // count=1, user mode
  g0[1] = lds_off;                                   // LDS byte address
  g0[2] = (unsigned)(ga & 0xFFFFFFFFu);              // global_addr lo
  g0[3] = (unsigned)((ga >> 32) & 0x1FFFFFFu) | (2u << 30);  // hi | type=2
  tdm_v8i g1;
  g1[0] = (int)((1u << 16)                           // data_size = 2 bytes
              | (1u << 20)                           // pad_enable
              | (6u << 22)                           // pad_interval: 512B
              | (3u << 25));                         // pad_amount: 4 dwords
  g1[1] = (int)(((unsigned)K1 & 0xFFFFu) << 16);     // tensor_dim0 lo16
  g1[2] = (int)(((unsigned)K1 >> 16) | (64u << 16)); // td0 hi | tensor_dim1 lo
  g1[3] = (int)((unsigned)K1 << 16);                 // td1 hi | tile_dim0=256
  g1[4] = (int)64u;                                  // tile_dim1=64, tile_dim2=0
  g1[5] = (int)(unsigned)K1;                         // tensor_dim0_stride lo32
  g1[6] = 0;
  g1[7] = 0;
  tdm_v4i z4 = {0, 0, 0, 0};
#if __clang_major__ >= 23
  tdm_v8i z8 = {0, 0, 0, 0, 0, 0, 0, 0};
  __builtin_amdgcn_tensor_load_to_lds(g0, g1, z4, z4, z8, 0);
#else
  __builtin_amdgcn_tensor_load_to_lds(g0, g1, z4, z4, 0);
#endif
}
#endif

__device__ __forceinline__ v16bf ld16(const __bf16* p16a, const __bf16* p16b) {
  v8bf lo = *(const v8bf*)p16a;
  v8bf hi = *(const v8bf*)p16b;
  return __builtin_shufflevector(lo, hi, 0, 1, 2, 3, 4, 5, 6, 7,
                                         8, 9, 10, 11, 12, 13, 14, 15);
}

__device__ __forceinline__ __bf16 spike_bf(float v) {
  return (v >= 1.0f) ? (__bf16)1.0f : (__bf16)0.0f;
}

// layer 1: S[M1][N1] = spike(Abuf @ Wc1). 8 waves/block share one TDM-staged
// B tile in LDS; A tile held in registers; K-loop is ds_load + wmma only.
__global__ __launch_bounds__(256)
void k_gemm1(const __bf16* __restrict__ A, const __bf16* __restrict__ WT,
             __bf16* __restrict__ S) {
  __shared__ __bf16 sB[64 * PITCH1];                 // 64 rows, 528B pitch
  int lane = threadIdx.x & 31, wave = threadIdx.x >> 5;
  int m0 = blockIdx.x * 128 + wave * 16;
  int n0 = blockIdx.y * 64;

#if USE_TDM
  if (wave == 0) {
    tdm_load_B1((unsigned)(size_t)(void*)sB, WT + (size_t)n0 * K1);
    __builtin_amdgcn_s_wait_tensorcnt(0);
  }
#else
  for (int i = threadIdx.x; i < 64 * 32; i += 256) { // padded manual copy
    int r = i >> 5, seg = i & 31;
    *(v8bf*)(sB + r * PITCH1 + seg * 8) =
        *(const v8bf*)(WT + ((size_t)n0 + r) * K1 + seg * 8);
  }
#endif

  // preload the wave's 16x256 A tile into registers (overlaps the TDM)
  const __bf16* arow = A + (size_t)(m0 + (lane & 15)) * K1 + ((lane >> 4) * 8);
  v16bf a[8];
#pragma unroll
  for (int i = 0; i < 8; ++i)
    a[i] = ld16(arow + i * 32, arow + i * 32 + 16);

  __syncthreads();

  const __bf16* brow = sB + (lane & 15) * PITCH1 + (lane >> 4) * 8;
  v8f acc0 = {}, acc1 = {}, acc2 = {}, acc3 = {};
#pragma unroll
  for (int k = 0; k < 8; ++k) {
    const __bf16* bk = brow + k * 32;
    v16bf b0 = ld16(bk, bk + 16);
    v16bf b1 = ld16(bk + 16 * PITCH1, bk + 16 * PITCH1 + 16);
    v16bf b2 = ld16(bk + 32 * PITCH1, bk + 32 * PITCH1 + 16);
    v16bf b3 = ld16(bk + 48 * PITCH1, bk + 48 * PITCH1 + 16);
    acc0 = __builtin_amdgcn_wmma_f32_16x16x32_bf16(false, a[k], false, b0, (short)0, acc0, false, false);
    acc1 = __builtin_amdgcn_wmma_f32_16x16x32_bf16(false, a[k], false, b1, (short)0, acc1, false, false);
    acc2 = __builtin_amdgcn_wmma_f32_16x16x32_bf16(false, a[k], false, b2, (short)0, acc2, false, false);
    acc3 = __builtin_amdgcn_wmma_f32_16x16x32_bf16(false, a[k], false, b3, (short)0, acc3, false, false);
  }

  int srow = m0 + ((lane >> 4) << 3);
  int scol = n0 + (lane & 15);
#pragma unroll
  for (int r = 0; r < 8; ++r) {
    size_t base = (size_t)(srow + r) * N1 + scol;
    S[base +  0] = spike_bf(acc0[r]);
    S[base + 16] = spike_bf(acc1[r]);
    S[base + 32] = spike_bf(acc2[r]);
    S[base + 48] = spike_bf(acc3[r]);
  }
}

// layer-2 input: B2[b] = [ s0[b] | mean_{j<5} s1[b*5+j] ]
__global__ __launch_bounds__(256)
void k_build_B2(const __bf16* __restrict__ S, __bf16* __restrict__ B2) {
  int wave = threadIdx.x >> 5, lane = threadIdx.x & 31;
  int b = blockIdx.x * 8 + wave;
  int c0 = lane * 16;                                 // 16 cols per lane
  const __bf16* s0row = S + (size_t)b * N1;
  __bf16* orow = B2 + (size_t)b * K2;
  *(v16bf*)(orow + c0) = *(const v16bf*)(s0row + c0);
  float sum[16];
#pragma unroll
  for (int e = 0; e < 16; ++e) sum[e] = 0.f;
#pragma unroll
  for (int j = 0; j < S1_; ++j) {
    v16bf v = *(const v16bf*)(S + (size_t)(B_SEEDS + b * S1_ + j) * N1 + c0);
#pragma unroll
    for (int e = 0; e < 16; ++e) sum[e] += (float)v[e];
  }
#pragma unroll
  for (int e = 0; e < 16; ++e)
    orow[H1_ + c0 + e] = (__bf16)(sum[e] * (1.0f / S1_));
}

// layer 2: out[M2][N2] = spike(B2 @ Wc2), f32 (direct-global WMMA)
__global__ __launch_bounds__(128)
void k_gemm2(const __bf16* __restrict__ B2, const __bf16* __restrict__ WT,
             float* __restrict__ outp) {
  int lane = threadIdx.x & 31, wave = threadIdx.x >> 5;
  int m0 = blockIdx.x * 64 + wave * 16;
  const __bf16* arow = B2 + (size_t)(m0 + (lane & 15)) * K2 + ((lane >> 4) * 16);
  const __bf16* wrow = WT + (size_t)(lane & 15) * K2 + (lane >> 4) * 16;
  v8f acc0 = {}, acc1 = {}, acc2 = {}, acc3 = {};
  for (int k0 = 0; k0 < K2; k0 += 32) {
    if (k0 + 32 < K2) __builtin_prefetch(arow + k0 + 32, 0, 1);
    v16bf a = ld16(arow + k0, arow + k0 + 8);
    v16bf b0 = *(const v16bf*)(wrow + k0);
    v16bf b1 = *(const v16bf*)(wrow + k0 + (size_t)16 * K2);
    v16bf b2 = *(const v16bf*)(wrow + k0 + (size_t)32 * K2);
    v16bf b3 = *(const v16bf*)(wrow + k0 + (size_t)48 * K2);
    acc0 = __builtin_amdgcn_wmma_f32_16x16x32_bf16(false, a, false, b0, (short)0, acc0, false, false);
    acc1 = __builtin_amdgcn_wmma_f32_16x16x32_bf16(false, a, false, b1, (short)0, acc1, false, false);
    acc2 = __builtin_amdgcn_wmma_f32_16x16x32_bf16(false, a, false, b2, (short)0, acc2, false, false);
    acc3 = __builtin_amdgcn_wmma_f32_16x16x32_bf16(false, a, false, b3, (short)0, acc3, false, false);
  }
  int srow = m0 + ((lane >> 4) << 3);
  int nc = lane & 15;
#pragma unroll
  for (int r = 0; r < 8; ++r) {
    size_t base = (size_t)(srow + r) * N2 + nc;
    outp[base +  0] = (acc0[r] >= 1.0f) ? 1.0f : 0.0f;
    outp[base + 16] = (acc1[r] >= 1.0f) ? 1.0f : 0.0f;
    outp[base + 32] = (acc2[r] >= 1.0f) ? 1.0f : 0.0f;
    outp[base + 48] = (acc3[r] >= 1.0f) ? 1.0f : 0.0f;
  }
}

// ---------------- temporal attention + MTG head --------------------------
__device__ __forceinline__ float pos_val(int t, int i) {
  float e = (float)(i & ~1) * (1.0f / (float)H2_);
  float angle = (float)t * expf(-e * 9.210340371976184f);  // 10000^-e
  return (i & 1) ? cosf(angle) : sinf(angle);
}

__global__ __launch_bounds__(256)
void k_attn(const float* __restrict__ outs,   // [C][T][B][H2] (c1 uses slots 0..3)
            const float* __restrict__ pw, const float* __restrict__ pb,
            const float* __restrict__ nw, const float* __restrict__ nb,
            const float* __restrict__ mtgw, const float* __restrict__ mtgb,
            float* __restrict__ out) {
  __shared__ float lds_emb[8][H2_];
  int wave = threadIdx.x >> 5, lane = threadIdx.x & 31;
  int b = blockIdx.x * 8 + wave;
  int d0 = lane * 2;
  float emb0 = 0.f, emb1 = 0.f;                       // channel-mean embedding
  for (int c = 0; c < C_; ++c) {
    int Tc = c ? (T_STEPS / 2) : T_STEPS;
    float wp0 = pw[c * H2_ + d0], wp1 = pw[c * H2_ + d0 + 1];
    float wn0 = nw[c * H2_ + d0], wn1 = nw[c * H2_ + d0 + 1];
    float sv0[T_STEPS], sv1[T_STEPS], scp[T_STEPS], scn[T_STEPS];
    for (int t = 0; t < Tc; ++t) {
      const float* row = outs + (((size_t)c * T_STEPS + t) * B_SEEDS + b) * H2_;
      float v0 = row[d0], v1 = row[d0 + 1];
      sv0[t] = v0; sv1[t] = v1;
      float p0 = pos_val(t, d0), p1 = pos_val(t, d0 + 1);
      float part_p = (v0 + p0) * wp0 + (v1 + p1) * wp1;
      float part_n = v0 * wn0 + v1 * wn1;
      for (int off = 16; off; off >>= 1) {
        part_p += __shfl_xor(part_p, off, 32);
        part_n += __shfl_xor(part_n, off, 32);
      }
      scp[t] = part_p + pb[c];
      scn[t] = part_n + nb[c];
    }
    float mp = scp[0], mn = scn[0];
    for (int t = 1; t < Tc; ++t) { mp = fmaxf(mp, scp[t]); mn = fmaxf(mn, scn[t]); }
    float sp = 0.f, sn = 0.f;
    for (int t = 0; t < Tc; ++t) { sp += expf(scp[t] - mp); sn += expf(scn[t] - mn); }
    float ep0 = 0.f, ep1 = 0.f, en0 = 0.f, en1 = 0.f;
    for (int t = 0; t < Tc; ++t) {
      float wtp = expf(scp[t] - mp) / sp;
      float wtn = expf(scn[t] - mn) / sn;
      ep0 += wtp * (sv0[t] + pos_val(t, d0));
      ep1 += wtp * (sv1[t] + pos_val(t, d0 + 1));
      en0 += wtn * sv0[t];
      en1 += wtn * sv1[t];
    }
    emb0 += (1.0f / C_) * (0.6f * ep0 + 0.4f * en0);
    emb1 += (1.0f / C_) * (0.6f * ep1 + 0.4f * en1);
  }
  lds_emb[wave][d0] = emb0;
  lds_emb[wave][d0 + 1] = emb1;
  __syncthreads();
  float r = mtgb[lane];
  for (int h = 0; h < H2_; ++h) r += lds_emb[wave][h] * mtgw[h * O_ + lane];
  out[(size_t)b * O_ + lane] = r;
}

// ------------------------------ launcher ---------------------------------
extern "C" void kernel_launch(void* const* d_in, const int* in_sizes, int n_in,
                              void* d_out, int out_size, void* d_ws, size_t ws_size,
                              hipStream_t stream) {
  const int*   nodes = (const int*)d_in[0];
  const int*   hop1  = (const int*)d_in[1];
  const int*   hop2  = (const int*)d_in[2];
  const float* x     = (const float*)d_in[3];
  const float* Ws1   = (const float*)d_in[4];
  const float* Wn1   = (const float*)d_in[5];
  const float* Ws2   = (const float*)d_in[6];
  const float* Wn2   = (const float*)d_in[7];
  const float* pw    = (const float*)d_in[8];
  const float* pb    = (const float*)d_in[9];
  const float* nw    = (const float*)d_in[10];
  const float* nb    = (const float*)d_in[11];
  const float* mtgw  = (const float*)d_in[12];
  const float* mtgb  = (const float*)d_in[13];

  // workspace carve (256B aligned regions)
  char* p = (char*)d_ws;
  auto carve = [&](size_t bytes) {
    char* r = p;
    p += (bytes + 255) & ~(size_t)255;
    return r;
  };
  __bf16* WT1  = (__bf16*)carve((size_t)C_ * N1 * K1 * 2);   // 512 KB
  __bf16* WT2  = (__bf16*)carve((size_t)C_ * N2 * K2 * 2);   // 256 KB
  __bf16* Abuf = (__bf16*)carve((size_t)M1 * K1 * 2);        // 12.6 MB
  __bf16* Sbuf = (__bf16*)carve((size_t)M1 * N1 * 2);        // 25.2 MB
  __bf16* B2   = (__bf16*)carve((size_t)M2 * K2 * 2);        // 8.4 MB
  float*  outs = (float*)carve((size_t)C_ * T_STEPS * B_SEEDS * H2_ * 4); // 16.8 MB
  (void)ws_size; (void)in_sizes; (void)n_in; (void)out_size;

  k_prep_w1<<<(C_ * N1 * K1 + 255) / 256, 256, 0, stream>>>(Ws1, Wn1, WT1);
  k_prep_w2<<<(C_ * N2 * K2 + 255) / 256, 256, 0, stream>>>(Ws2, Wn2, WT2);

  for (int t = 0; t < T_STEPS; ++t) {
    k_build_A<<<M1 / 8, 256, 0, stream>>>(
        x + (size_t)t * N_NODES * F_DIM, nodes,
        hop1 + (size_t)t * B_SEEDS * S1_,
        hop2 + (size_t)t * B_SEEDS * S1_ * S2_, Abuf);

    // channel 0 (all timesteps)
    {
      dim3 g(M1 / 128, N1 / 64);
      k_gemm1<<<g, 256, 0, stream>>>(Abuf, WT1, Sbuf);
      k_build_B2<<<B_SEEDS / 8, 256, 0, stream>>>(Sbuf, B2);
      k_gemm2<<<M2 / 64, 128, 0, stream>>>(
          B2, WT2, outs + ((size_t)0 * T_STEPS + t) * B_SEEDS * H2_);
    }
    // channel 1 (even timesteps -> slot t/2)
    if ((t & 1) == 0) {
      dim3 g(M1 / 128, N1 / 64);
      k_gemm1<<<g, 256, 0, stream>>>(Abuf, WT1 + (size_t)N1 * K1, Sbuf);
      k_build_B2<<<B_SEEDS / 8, 256, 0, stream>>>(Sbuf, B2);
      k_gemm2<<<M2 / 64, 128, 0, stream>>>(
          B2, WT2 + (size_t)N2 * K2,
          outs + ((size_t)1 * T_STEPS + (t >> 1)) * B_SEEDS * H2_);
    }
  }

  k_attn<<<B_SEEDS / 8, 256, 0, stream>>>(outs, pw, pb, nw, nb, mtgw, mtgb,
                                          (float*)d_out);
}
